// co_attention_22213570855624
// MI455X (gfx1250) — matrix operand order repaired and checked by reading, not compiled, not run on hardware
//
#include <hip/hip_runtime.h>
#include <hip/hip_bf16.h>

// Co-attention for MI455X (gfx1250, wave32, WMMA + TDM).
// t:[B,N,D] f32, f:[B,N,D] f32, W:[D,D] f32 -> t_out,f_out:[B,D,N] f32.
// All matmuls run on v_wmma_f32_16x16x32_bf16 (bf16 operands, f32 accum).
// S (B x N x N) is never materialized: recomputed in 3 WMMA passes
// (max stats, sum stats, apply+accumulate) to stay inside L2 / small ws.
// final_kernel stages the f m-tile into LDS once per workgroup with the
// Tensor Data Mover (padded rows to dodge bank conflicts).

#define B_ 32
#define N_ 1024
#define D_ 256

typedef __attribute__((ext_vector_type(16))) __bf16          v16bf;
typedef __attribute__((ext_vector_type(8)))  float           v8f;
typedef __attribute__((ext_vector_type(8)))  unsigned short  us8;
typedef __attribute__((ext_vector_type(4)))  unsigned int    v4u;
typedef __attribute__((ext_vector_type(8)))  int             v8i;
typedef __attribute__((ext_vector_type(4)))  int             v4i;

union AB32 { us8 h[2]; v16bf v; };

__device__ __forceinline__ unsigned short f2bf(float f) {
  unsigned u = __builtin_bit_cast(unsigned, f);
  u += 0x7FFFu + ((u >> 16) & 1u);           // round-to-nearest-even
  return (unsigned short)(u >> 16);
}
// monotonic float<->uint mapping for atomicMax on floats
__device__ __forceinline__ unsigned fenc(float f) {
  unsigned u = __builtin_bit_cast(unsigned, f);
  return u ^ ((u >> 31) ? 0xFFFFFFFFu : 0x80000000u);
}
__device__ __forceinline__ float fdec(unsigned u) {
  unsigned b = u ^ ((u >> 31) ? 0x80000000u : 0xFFFFFFFFu);
  return __builtin_bit_cast(float, b);
}

// A-matrix 16x32 bf16 fragment: lane row = lane&15;
// half=0 lanes hold K={0..7,16..23}, half=1 lanes hold K={8..15,24..31}.
__device__ __forceinline__ v16bf fragA(const unsigned short* p, int half) {
  AB32 u;
  u.h[0] = *(const us8*)(p + half * 8);
  u.h[1] = *(const us8*)(p + 16 + half * 8);
  return u.v;
}
// B-matrix 32x16 bf16 fragment: lane col = lane&15;
// half=0 lanes hold K={0..7,16..23}, half=1 lanes hold K={8..15,24..31}.
__device__ __forceinline__ v16bf fragB(const unsigned short* p, int half) {
  AB32 u;
  u.h[0] = *(const us8*)(p + half * 8);
  u.h[1] = *(const us8*)(p + 16 + half * 8);
  return u.v;
}

// 16x16 f32 tile = rowA(16xD_) x rowB(D_x16), K-loop of 8 WMMAs.
// aRow/bRow are per-lane row pointers (row = lane&15 of the tile).
template <typename PtrB>
__device__ __forceinline__ v8f mm16x16(const unsigned short* aRow,
                                       PtrB bRow, int half) {
  v8f acc = {0.f, 0.f, 0.f, 0.f, 0.f, 0.f, 0.f, 0.f};
#pragma unroll
  for (int kc = 0; kc < D_; kc += 32) {
    v16bf a = fragA(aRow + kc, half);
    v16bf b = fragB(bRow + kc, half);
    acc = __builtin_amdgcn_wmma_f32_16x16x32_bf16(
        false, a, false, b, (short)0, acc, false, false);
  }
  return acc;
}

// ---------- kernel 1: f32 -> bf16 convert (+LDS-tiled transpose) ----------
__global__ void convtrans_kernel(const float* __restrict__ src,
                                 unsigned short* __restrict__ outS,
                                 unsigned short* __restrict__ outT,
                                 int rows, int cols) {
  __shared__ __align__(16) unsigned short tile[32][33];
  int tiles_c = cols >> 5;
  int tiles_per_b = (rows >> 5) * tiles_c;
  int bi  = blockIdx.x / tiles_per_b;
  int tid = blockIdx.x % tiles_per_b;
  int r0 = (tid / tiles_c) << 5;
  int c0 = (tid % tiles_c) << 5;
  int tx = threadIdx.x & 31, ty = threadIdx.x >> 5;
  size_t base = (size_t)bi * rows * cols;
#pragma unroll
  for (int i = 0; i < 4; ++i) {
    int r = r0 + ty + 8 * i;
    unsigned short v = f2bf(src[base + (size_t)r * cols + c0 + tx]);
    tile[ty + 8 * i][tx] = v;
    if (outS) outS[base + (size_t)r * cols + c0 + tx] = v;
  }
  __syncthreads();
#pragma unroll
  for (int i = 0; i < 4; ++i) {
    int c = c0 + ty + 8 * i;
    outT[base + (size_t)c * rows + r0 + tx] = tile[tx][ty + 8 * i];
  }
}

// ---------- kernel 2: zero softmax statistics ----------
__global__ void init_stats_kernel(unsigned* __restrict__ p, int n) {
  int i = blockIdx.x * blockDim.x + threadIdx.x;
  if (i < n) p[i] = 0u;   // 0 == identity for fenc-max; 0.0f for sums
}

// ---------- kernel 3: tW = t @ W (bf16 out) ----------
__global__ void tw_kernel(const unsigned short* __restrict__ t_bf,
                          const unsigned short* __restrict__ Wt,
                          unsigned short* __restrict__ tW) {
  int lane = threadIdx.x & 31;
  int wid = blockIdx.x * (blockDim.x >> 5) + (threadIdx.x >> 5);
  int c = lane & 15, half = lane >> 4;
  int dt = wid & 15;          // D/16 column tiles
  int nt = wid >> 4;          // B*N/16 row tiles
  int b = nt >> 6;
  int n0 = (nt & 63) << 4;
  int e0 = dt << 4;
  const unsigned short* aRow = t_bf + (size_t)(b * N_ + n0 + c) * D_;
  const unsigned short* bRow = Wt + (size_t)(e0 + c) * D_;   // Wt[e][d]=W[d][e]
  v8f acc = mm16x16(aRow, bRow, half);
#pragma unroll
  for (int j = 0; j < 8; ++j) {   // D-frag: row = j + 8*half, col = c
    tW[(size_t)(b * N_ + n0 + j + 8 * half) * D_ + e0 + c] = f2bf(acc[j]);
  }
}

// ---------- kernel 4: S tile -> row/col max (pass 1) ----------
__global__ void smax_kernel(const unsigned short* __restrict__ tW,
                            const unsigned short* __restrict__ f_bf,
                            unsigned* __restrict__ rowmax_u,
                            unsigned* __restrict__ colmax_u) {
  int lane = threadIdx.x & 31;
  int wid = blockIdx.x * (blockDim.x >> 5) + (threadIdx.x >> 5);
  int c = lane & 15, half = lane >> 4;
  int b = wid >> 12;
  int n0 = ((wid >> 6) & 63) << 4;
  int m0 = (wid & 63) << 4;
  const unsigned short* aRow = tW + (size_t)(b * N_ + n0 + c) * D_;
  const unsigned short* bRow = f_bf + (size_t)(b * N_ + m0 + c) * D_;
  v8f acc = mm16x16(aRow, bRow, half);
  // column max (reduce over n = VGPRs + cross-half)
  float cv = acc[0];
#pragma unroll
  for (int j = 1; j < 8; ++j) cv = fmaxf(cv, acc[j]);
  cv = fmaxf(cv, __shfl_xor(cv, 16, 32));
  if (lane < 16) atomicMax(&colmax_u[b * N_ + m0 + lane], fenc(cv));
  // row max (reduce over m = 16 lanes of a half-wave)
  float rv[8];
#pragma unroll
  for (int j = 0; j < 8; ++j) {
    float v = acc[j];
    v = fmaxf(v, __shfl_xor(v, 1, 32));
    v = fmaxf(v, __shfl_xor(v, 2, 32));
    v = fmaxf(v, __shfl_xor(v, 4, 32));
    v = fmaxf(v, __shfl_xor(v, 8, 32));
    rv[j] = v;
  }
  if (c == 0) {
#pragma unroll
    for (int j = 0; j < 8; ++j)
      atomicMax(&rowmax_u[b * N_ + n0 + 8 * half + j], fenc(rv[j]));
  }
}

// ---------- kernel 5: S tile -> row/col exp-sums (pass 2) ----------
__global__ void ssum_kernel(const unsigned short* __restrict__ tW,
                            const unsigned short* __restrict__ f_bf,
                            const unsigned* __restrict__ rowmax_u,
                            const unsigned* __restrict__ colmax_u,
                            float* __restrict__ rowsum,
                            float* __restrict__ colsum) {
  int lane = threadIdx.x & 31;
  int wid = blockIdx.x * (blockDim.x >> 5) + (threadIdx.x >> 5);
  int c = lane & 15, half = lane >> 4;
  int b = wid >> 12;
  int n0 = ((wid >> 6) & 63) << 4;
  int m0 = (wid & 63) << 4;
  const unsigned short* aRow = tW + (size_t)(b * N_ + n0 + c) * D_;
  const unsigned short* bRow = f_bf + (size_t)(b * N_ + m0 + c) * D_;
  v8f acc = mm16x16(aRow, bRow, half);
  float cm = fdec(colmax_u[b * N_ + m0 + c]);
  float colacc = 0.f;
  float rowe[8];
#pragma unroll
  for (int j = 0; j < 8; ++j) {
    float rm = fdec(rowmax_u[b * N_ + n0 + 8 * half + j]);
    rowe[j] = __expf(acc[j] - rm);
    colacc += __expf(acc[j] - cm);
  }
  colacc += __shfl_xor(colacc, 16, 32);
  if (lane < 16) atomicAdd(&colsum[b * N_ + m0 + lane], colacc);
#pragma unroll
  for (int j = 0; j < 8; ++j) {
    float v = rowe[j];
    v += __shfl_xor(v, 1, 32);
    v += __shfl_xor(v, 2, 32);
    v += __shfl_xor(v, 4, 32);
    v += __shfl_xor(v, 8, 32);
    rowe[j] = v;
  }
  if (c == 0) {
#pragma unroll
    for (int j = 0; j < 8; ++j)
      atomicAdd(&rowsum[b * N_ + n0 + 8 * half + j], rowe[j]);
  }
}

// ---------- kernel 6: recompute S, normalize, accumulate outputs ----------
// One workgroup (8 waves) per (b, m-tile). TDM stages the f m-tile
// (16 x 256 bf16) into LDS once, with pad_interval=128DW/pad_amount=4DW so
// rows land at stride 264 ushorts (bank-conflict-free column reads). Each
// wave produces 32 rows of both weight matrices into LDS (B-fragment
// order), then wave w accumulates d-tiles {2w,2w+1} of t_out and f_out.
__global__ void __launch_bounds__(256) final_kernel(
    const unsigned short* __restrict__ tW,
    const unsigned short* __restrict__ f_bf,
    const unsigned short* __restrict__ tT,
    const unsigned short* __restrict__ fT,
    const unsigned* __restrict__ rowmax_u,
    const unsigned* __restrict__ colmax_u,
    const float* __restrict__ rowsum,
    const float* __restrict__ colsum,
    float* __restrict__ t_out, float* __restrict__ f_out) {
  __shared__ __align__(16) unsigned short wgtT[16][264];  // [m-col][n-local]
  __shared__ __align__(16) unsigned short wgtF[16][264];
  __shared__ __align__(16) unsigned short fTile[16][264]; // TDM-padded rows
  int lane = threadIdx.x & 31;
  int wave = threadIdx.x >> 5;               // 0..7
  int c = lane & 15, half = lane >> 4;
  int b = blockIdx.x >> 6;
  int m0 = (blockIdx.x & 63) << 4;

  // ---- stage f[m0:m0+16, 0:256] into LDS via Tensor Data Mover ----
  const unsigned short* fSrc = f_bf + (size_t)(b * N_ + m0) * D_;
#if __has_builtin(__builtin_amdgcn_tensor_load_to_lds)
  if (threadIdx.x < 32) {                    // wave 0 issues one TDM op
    unsigned lds_off = (unsigned)(size_t)(&fTile[0][0]);
    unsigned long long ga = (unsigned long long)(size_t)fSrc;
    v4u g0 = {0u, 0u, 0u, 0u};
    g0[0] = 1u;                              // count=1 (valid), user mode
    g0[1] = lds_off;                         // lds_addr
    g0[2] = (unsigned)(ga & 0xFFFFFFFFu);    // global_addr[31:0]
    g0[3] = (unsigned)((ga >> 32) & 0x01FFFFFFu) | (2u << 30);  // addr+type=2
    v8i g1 = {0, 0, 0, 0, 0, 0, 0, 0};
    // data_size=2B (1<<16) | pad_enable (1<<20) | pad_interval=128DW (6<<22)
    // | pad_amount=4DW (3<<25)
    g1[0] = (int)((1u << 16) | (1u << 20) | (6u << 22) | (3u << 25));
    g1[1] = (int)(256u << 16);               // tensor_dim0 = 256
    g1[2] = (int)(16u << 16);                // tensor_dim1 = 16
    g1[3] = (int)(256u << 16);               // tile_dim0 = 256
    g1[4] = 16;                              // tile_dim1 = 16
    g1[5] = 256;                             // tensor_dim0_stride = 256
    v4i gz4 = {0, 0, 0, 0};
    v8i gz8 = {0, 0, 0, 0, 0, 0, 0, 0};
    __builtin_amdgcn_tensor_load_to_lds(g0, g1, gz4, gz4, gz8, 0);
#if __has_builtin(__builtin_amdgcn_s_wait_tensorcnt)
    __builtin_amdgcn_s_wait_tensorcnt(0);
#else
    asm volatile("s_wait_tensorcnt 0x0" ::: "memory");
#endif
  }
#else
  for (int i = threadIdx.x; i < 16 * 256; i += 256)
    fTile[i >> 8][i & 255] = fSrc[(size_t)(i >> 8) * D_ + (i & 255)];
#endif
  __syncthreads();

  float cmax = fdec(colmax_u[b * N_ + m0 + c]);
  float cinv = 1.0f / colsum[b * N_ + m0 + c];
  const unsigned short* bRow = &fTile[c][0];   // LDS-resident B rows

  int d0a = (2 * wave) << 4, d0b = (2 * wave + 1) << 4;
  const unsigned short* aT0 = tT + (size_t)(b * D_ + d0a + c) * N_;
  const unsigned short* aT1 = tT + (size_t)(b * D_ + d0b + c) * N_;
  const unsigned short* aF0 = fT + (size_t)(b * D_ + d0a + c) * N_;
  const unsigned short* aF1 = fT + (size_t)(b * D_ + d0b + c) * N_;

  v8f acc_t0 = {0.f,0.f,0.f,0.f,0.f,0.f,0.f,0.f};
  v8f acc_t1 = acc_t0, acc_f0 = acc_t0, acc_f1 = acc_t0;

  for (int nb0 = 0; nb0 < N_; nb0 += 256) {
    // phase 1: wave computes S rows [nb0+32w, nb0+32w+32), normalizes both ways
#pragma unroll
    for (int tile = 0; tile < 2; ++tile) {
      int nbase = nb0 + 32 * wave + 16 * tile;
      const unsigned short* aRow = tW + (size_t)(b * N_ + nbase + c) * D_;
      v8f s = mm16x16(aRow, bRow, half);
      union { unsigned short u[8]; us8 v; } pt, pf;
#pragma unroll
      for (int j = 0; j < 8; ++j) {
        int n = nbase + 8 * half + j;                 // D-frag row
        float rm = fdec(rowmax_u[b * N_ + n]);
        float ri = 1.0f / rowsum[b * N_ + n];
        pt.u[j] = f2bf(__expf(s[j] - cmax) * cinv);   // softmax over n
        pf.u[j] = f2bf(__expf(s[j] - rm) * ri);       // softmax over m
      }
      int off = 32 * wave + 16 * tile + 8 * half;
      *(us8*)&wgtT[c][off] = pt.v;
      *(us8*)&wgtF[c][off] = pf.v;
    }
    __syncthreads();
    // prefetch next n-block's A operands while we consume this one
    if (nb0 + 256 < N_) {
      __builtin_prefetch(aT0 + nb0 + 256, 0, 0);
      __builtin_prefetch(aF0 + nb0 + 256, 0, 0);
    }
    // phase 2: accumulate outputs; weights read from LDS as B fragments
#pragma unroll
    for (int kb = 0; kb < 256; kb += 32) {
      v16bf bT = fragB(&wgtT[c][kb], half);
      v16bf bF = fragB(&wgtF[c][kb], half);
      v16bf a;
      a = fragA(aT0 + nb0 + kb, half);
      acc_t0 = __builtin_amdgcn_wmma_f32_16x16x32_bf16(false, a, false, bT,
                                                       (short)0, acc_t0, false, false);
      a = fragA(aT1 + nb0 + kb, half);
      acc_t1 = __builtin_amdgcn_wmma_f32_16x16x32_bf16(false, a, false, bT,
                                                       (short)0, acc_t1, false, false);
      a = fragA(aF0 + nb0 + kb, half);
      acc_f0 = __builtin_amdgcn_wmma_f32_16x16x32_bf16(false, a, false, bF,
                                                       (short)0, acc_f0, false, false);
      a = fragA(aF1 + nb0 + kb, half);
      acc_f1 = __builtin_amdgcn_wmma_f32_16x16x32_bf16(false, a, false, bF,
                                                       (short)0, acc_f1, false, false);
    }
    __syncthreads();
  }
#pragma unroll
  for (int j = 0; j < 8; ++j) {
    int dA = d0a + j + 8 * half, dB = d0b + j + 8 * half;
    t_out[(size_t)(b * D_ + dA) * N_ + m0 + c] = acc_t0[j];
    t_out[(size_t)(b * D_ + dB) * N_ + m0 + c] = acc_t1[j];
    f_out[(size_t)(b * D_ + dA) * N_ + m0 + c] = acc_f0[j];
    f_out[(size_t)(b * D_ + dB) * N_ + m0 + c] = acc_f1[j];
  }
}

extern "C" void kernel_launch(void* const* d_in, const int* in_sizes, int n_in,
                              void* d_out, int out_size, void* d_ws, size_t ws_size,
                              hipStream_t stream) {
  (void)in_sizes; (void)n_in; (void)out_size; (void)ws_size;
  const float* t = (const float*)d_in[0];
  const float* f = (const float*)d_in[1];
  const float* W = (const float*)d_in[2];
  const size_t SZ = (size_t)B_ * N_ * D_;       // 8,388,608 elements
  unsigned short* wsu  = (unsigned short*)d_ws; // ws usage ~84.5 MB
  unsigned short* t_bf = wsu;
  unsigned short* f_bf = wsu + SZ;
  unsigned short* tT   = wsu + 2 * SZ;
  unsigned short* fT   = wsu + 3 * SZ;
  unsigned short* tW   = wsu + 4 * SZ;
  unsigned short* Wt   = wsu + 5 * SZ;
  unsigned* rowmax_u = (unsigned*)(wsu + 5 * SZ + (size_t)D_ * D_);
  unsigned* colmax_u = rowmax_u + B_ * N_;
  float* rowsum = (float*)(colmax_u + B_ * N_);
  float* colsum = rowsum + B_ * N_;
  float* t_out = (float*)d_out;
  float* f_out = t_out + (size_t)B_ * D_ * N_;

  convtrans_kernel<<<B_ * (N_ / 32) * (D_ / 32), 256, 0, stream>>>(t, t_bf, tT, N_, D_);
  convtrans_kernel<<<B_ * (N_ / 32) * (D_ / 32), 256, 0, stream>>>(f, f_bf, fT, N_, D_);
  convtrans_kernel<<<(D_ / 32) * (D_ / 32), 256, 0, stream>>>(W, nullptr, Wt, D_, D_);
  init_stats_kernel<<<(4 * B_ * N_) / 256, 256, 0, stream>>>(rowmax_u, 4 * B_ * N_);
  tw_kernel<<<(B_ * (N_ / 16) * (D_ / 16)) / 8, 256, 0, stream>>>(t_bf, Wt, tW);
  smax_kernel<<<(B_ * 64 * 64) / 8, 256, 0, stream>>>(tW, f_bf, rowmax_u, colmax_u);
  ssum_kernel<<<(B_ * 64 * 64) / 8, 256, 0, stream>>>(tW, f_bf, rowmax_u, colmax_u,
                                                      rowsum, colsum);
  final_kernel<<<B_ * 64, 256, 0, stream>>>(tW, f_bf, tT, fT, rowmax_u, colmax_u,
                                            rowsum, colsum, t_out, f_out);
}
